// Transformer_24575802868476
// MI455X (gfx1250) — matrix operand order repaired and checked by reading, not compiled, not run on hardware
//
#include <hip/hip_runtime.h>

// ---------------------------------------------------------------------------
// Transformer block (softmax-free attention, reassociated) on MI455X/gfx1250.
// bf16 WMMA 16x16x32, f32 accumulate. LayerNorm/residuals in fp32.
// Deterministic: per-block KtV partials reduced in fixed order (no fp atomics).
// v3: + global_prefetch of next row-tile in kernel 1 (overlap HBM latency
//     with WMMA compute). Otherwise identical to v2 (good histogram/asm).
// ---------------------------------------------------------------------------

#define N_ROWS   131072
#define D_MODEL  128
#define D_FF     512
#define T_STEPS  4
#define ROWS_PER_TILE   128                 // 8 waves x 16 rows
#define TILES_PER_BLOCK 4
#define NBLOCKS1 (N_ROWS / (ROWS_PER_TILE * TILES_PER_BLOCK))   // 256
#define NBLOCKS3 (N_ROWS / ROWS_PER_TILE)                       // 1024

typedef __bf16 bf16_t;
typedef bf16_t v16bf __attribute__((ext_vector_type(16)));
typedef bf16_t v8bf  __attribute__((ext_vector_type(8)));
typedef bf16_t v4bf  __attribute__((ext_vector_type(4)));
typedef float  v8f   __attribute__((ext_vector_type(8)));

__device__ __forceinline__ v8f wmma_bf16(v16bf a, v16bf b, v8f c) {
  // D(f32 16x16) = A(bf16 16x32) * B(bf16 32x16) + C
  return __builtin_amdgcn_wmma_f32_16x16x32_bf16(false, a, false, b,
                                                 (short)0, c, false, false);
}

// Wave-local LDS ordering: DS ops from one wave complete in order; wait for
// outstanding DS ops so cross-lane (same-wave) LDS RAW is safe. Compiler
// reordering blocked by the memory clobber. No inter-wave sync implied.
__device__ __forceinline__ void wave_lds_fence() {
  asm volatile("s_wait_dscnt 0x0" ::: "memory");
}

// A-fragment (16x32 bf16). base -> element (row0, k0), row-major, 'stride'
// elements/row (stride, k0 multiples of 8). ISA layout: lane(m=l&15,
// half=l>>4); elem j<8 -> K = 8*half + j; elem j>=8 -> K = 16 + 8*half + j-8.
__device__ __forceinline__ v16bf load_a_frag(const bf16_t* base, int stride) {
  const int lane = threadIdx.x & 31;
  const int m    = lane & 15;
  const int hl   = lane >> 4;
  const bf16_t* p = base + m * stride + (hl << 3);
  v8bf lo = *(const v8bf*)(p);
  v8bf hi = *(const v8bf*)(p + 16);
  return __builtin_shufflevector(lo, hi, 0,1,2,3,4,5,6,7,8,9,10,11,12,13,14,15);
}

// B-fragment (32x16 bf16): lane l supplies col n0+(l&15); elem j<8 -> K = 8*half + j;
// elem j>=8 -> K = 16 + 8*half + j-8 (same K order as A).
__device__ __forceinline__ v16bf load_b_frag(const bf16_t* base, int stride) {
  const int lane = threadIdx.x & 31;
  const int n    = lane & 15;
  const int hl   = lane >> 4;
  v16bf b;
  for (int s = 0; s < 16; ++s) {
    const int k = (s < 8) ? (8 * hl + s) : (16 + 8 * hl + (s - 8));
    b[s] = base[k * stride + n];
  }
  return b;
}

// wave32 all-reduce of sum/sumsq for a 128-wide row (4 elems per lane).
__device__ __forceinline__ void wave_stats(float4 vv, float& mean, float& inv) {
  float s  = vv.x + vv.y + vv.z + vv.w;
  float s2 = vv.x*vv.x + vv.y*vv.y + vv.z*vv.z + vv.w*vv.w;
  for (int o = 16; o > 0; o >>= 1) {
    s  += __shfl_xor(s,  o, 32);
    s2 += __shfl_xor(s2, o, 32);
  }
  mean = s * (1.0f / 128.0f);
  float var = s2 * (1.0f / 128.0f) - mean * mean;
  inv = rsqrtf(var + 1e-5f);
}

// ---------------------------------------------------------------------------
__global__ void k_cvt_bf16(const float* __restrict__ src,
                           bf16_t* __restrict__ dst, int n) {
  int i = blockIdx.x * blockDim.x + threadIdx.x;
  if (i < n) dst[i] = (bf16_t)src[i];
}

// ---------------------------------------------------------------------------
// Kernel 1: for 4 row-tiles: h=LN1(x); K,V=h@{Kw,Vw}; accumulate partial
// K^T V (per wave: d1 slice [r0,r0+16) x 128) in registers; one store.
__global__ void __launch_bounds__(256) k_ln1_ktv(
    const float* __restrict__ x,
    const bf16_t* __restrict__ Kw, const bf16_t* __restrict__ Vw,
    const float* __restrict__ g1, const float* __restrict__ b1,
    float* __restrict__ partials)
{
  __shared__ __align__(32) bf16_t ls_h [128 * 128];
  __shared__ __align__(32) bf16_t ls_kt[128 * 136];  // [d1][row], padded
  __shared__ __align__(32) bf16_t ls_v [128 * 128];  // [row][d2]
  const int wave = threadIdx.x >> 5;
  const int lane = threadIdx.x & 31;
  const int r0   = wave * 16;
  const int hl   = lane >> 4, nl = lane & 15;
  const int mb   = r0 + hl * 8;
  const v8f vzero = {0.f,0.f,0.f,0.f,0.f,0.f,0.f,0.f};

  v8f acc[8];
  for (int n = 0; n < 8; ++n) acc[n] = vzero;

  // Prefetch first tile (each thread covers one 256B line of the 64KB tile).
  {
    const size_t base0 = (size_t)blockIdx.x * TILES_PER_BLOCK * ROWS_PER_TILE;
    __builtin_prefetch(x + base0 * 128 + (size_t)threadIdx.x * 64, 0, 3);
  }

  for (int tile = 0; tile < TILES_PER_BLOCK; ++tile) {
    const size_t rowBase =
        ((size_t)blockIdx.x * TILES_PER_BLOCK + tile) * ROWS_PER_TILE;

    // Prefetch next tile's x while this tile computes.
    if (tile + 1 < TILES_PER_BLOCK)
      __builtin_prefetch(x + (rowBase + ROWS_PER_TILE) * 128 +
                             (size_t)threadIdx.x * 64, 0, 3);

    // ---- LN1 -> ls_h (bf16, wave-local rows) ----
    for (int r = 0; r < 16; ++r) {
      float4 vv = ((const float4*)(x + (rowBase + r0 + r) * 128))[lane];
      float mean, inv; wave_stats(vv, mean, inv);
      const int c0 = lane * 4;
      v4bf h;
      h[0] = (bf16_t)((vv.x - mean) * inv * g1[c0+0] + b1[c0+0]);
      h[1] = (bf16_t)((vv.y - mean) * inv * g1[c0+1] + b1[c0+1]);
      h[2] = (bf16_t)((vv.z - mean) * inv * g1[c0+2] + b1[c0+2]);
      h[3] = (bf16_t)((vv.w - mean) * inv * g1[c0+3] + b1[c0+3]);
      *(v4bf*)&ls_h[(r0 + r) * 128 + c0] = h;
    }
    wave_lds_fence();

    v16bf a[4];
    for (int kk = 0; kk < 4; ++kk) a[kk] = load_a_frag(&ls_h[r0*128 + kk*32], 128);

    for (int n = 0; n < 8; ++n) {          // K -> ls_kt (transposed)
      v8f c = vzero;
      for (int kk = 0; kk < 4; ++kk)
        c = wmma_bf16(a[kk], load_b_frag(Kw + kk*32*128 + n*16, 128), c);
      const int nn = n*16 + nl;
      for (int v = 0; v < 8; ++v)
        ls_kt[nn * 136 + (mb + v)] = (bf16_t)c[v];
    }
    for (int n = 0; n < 8; ++n) {          // V -> ls_v
      v8f c = vzero;
      for (int kk = 0; kk < 4; ++kk)
        c = wmma_bf16(a[kk], load_b_frag(Vw + kk*32*128 + n*16, 128), c);
      const int nn = n*16 + nl;
      for (int v = 0; v < 8; ++v)
        ls_v[(mb + v) * 128 + nn] = (bf16_t)c[v];
    }
    __syncthreads();                       // K,V cross-wave from here

    v16bf at[4];
    for (int kk = 0; kk < 4; ++kk)
      at[kk] = load_a_frag(&ls_kt[r0*136 + kk*32], 136);
    for (int n = 0; n < 8; ++n)
      for (int kk = 0; kk < 4; ++kk)
        acc[n] = wmma_bf16(at[kk], load_b_frag(&ls_v[kk*32*128 + n*16], 128),
                           acc[n]);
    __syncthreads();                       // reads done before next-tile writes
  }

  float* pout = partials + (size_t)blockIdx.x * (128 * 128);
  for (int n = 0; n < 8; ++n) {
    const int nn = n*16 + nl;
    for (int v = 0; v < 8; ++v)
      pout[(mb + v) * 128 + nn] = acc[n][v];
  }
}

// ---------------------------------------------------------------------------
// Kernel 2: fixed-order reduction of 256 partials -> bf16 KtV (128x128).
__global__ void __launch_bounds__(256) k_ktv_reduce(
    const float* __restrict__ partials, bf16_t* __restrict__ ktv)
{
  const int i4 = (blockIdx.x * 256 + threadIdx.x) * 4;   // 16 blocks
  float4 s = {0.f, 0.f, 0.f, 0.f};
  for (int p = 0; p < NBLOCKS1; ++p) {
    float4 v = *(const float4*)&partials[(size_t)p * (128*128) + i4];
    s.x += v.x; s.y += v.y; s.z += v.z; s.w += v.w;
  }
  v4bf o; o[0]=(bf16_t)s.x; o[1]=(bf16_t)s.y; o[2]=(bf16_t)s.z; o[3]=(bf16_t)s.w;
  *(v4bf*)&ktv[i4] = o;
}

// ---------------------------------------------------------------------------
// Kernel 3: h1=LN1(x); Q=h1@Qw (recomputed, staged C->A via LDS);
// x += Q@KtV; h2=LN2(x); x += relu(h2@w1+b1)@w2 + b2; store x.
// All LDS buffers are wave-partitioned -> wave fences only, no barriers.
__global__ void __launch_bounds__(256) k_attn_ffn(
    const float* __restrict__ xin, float* __restrict__ xout,
    const bf16_t* __restrict__ Qw, const bf16_t* __restrict__ ktv,
    const float* __restrict__ g1, const float* __restrict__ b1,
    const float* __restrict__ g2, const float* __restrict__ b2,
    const bf16_t* __restrict__ w1, const float* __restrict__ bias1,
    const bf16_t* __restrict__ w2, const float* __restrict__ bias2)
{
  __shared__ __align__(32) float  ls_x[128 * 128];
  __shared__ __align__(32) bf16_t ls_h[128 * 128];    // h1 -> Q -> h2 staging
  __shared__ __align__(32) bf16_t ls_u[8][16 * 32];   // per-wave FF chunk
  const int wave = threadIdx.x >> 5;
  const int lane = threadIdx.x & 31;
  const int r0   = wave * 16;
  const int hl   = lane >> 4, nl = lane & 15;
  const int mb   = r0 + hl * 8;
  const size_t rowBase = (size_t)blockIdx.x * ROWS_PER_TILE;
  const v8f vzero = {0.f,0.f,0.f,0.f,0.f,0.f,0.f,0.f};

  // ---- load x tile + LN1 in one pass ----
  for (int r = 0; r < 16; ++r) {
    float4 vv = ((const float4*)(xin + (rowBase + r0 + r) * 128))[lane];
    *(float4*)&ls_x[(r0 + r) * 128 + lane * 4] = vv;
    float mean, inv; wave_stats(vv, mean, inv);
    const int c0 = lane * 4;
    v4bf h;
    h[0] = (bf16_t)((vv.x - mean) * inv * g1[c0+0] + b1[c0+0]);
    h[1] = (bf16_t)((vv.y - mean) * inv * g1[c0+1] + b1[c0+1]);
    h[2] = (bf16_t)((vv.z - mean) * inv * g1[c0+2] + b1[c0+2]);
    h[3] = (bf16_t)((vv.w - mean) * inv * g1[c0+3] + b1[c0+3]);
    *(v4bf*)&ls_h[(r0 + r) * 128 + c0] = h;
  }
  wave_lds_fence();

  // ---- Q = h1 @ Qw ; stage C-layout -> ls_h (overwrites h1) ----
  v16bf ah[4];
  for (int kk = 0; kk < 4; ++kk) ah[kk] = load_a_frag(&ls_h[r0*128 + kk*32], 128);
  wave_lds_fence();                         // h1 reads complete before overwrite
  for (int n = 0; n < 8; ++n) {
    v8f c = vzero;
    for (int kk = 0; kk < 4; ++kk)
      c = wmma_bf16(ah[kk], load_b_frag(Qw + kk*32*128 + n*16, 128), c);
    const int nn = n*16 + nl;
    for (int v = 0; v < 8; ++v)
      ls_h[(mb + v) * 128 + nn] = (bf16_t)c[v];
  }
  wave_lds_fence();

  // ---- x += Q @ KtV ----
  v16bf aq[4];
  for (int kk = 0; kk < 4; ++kk) aq[kk] = load_a_frag(&ls_h[r0*128 + kk*32], 128);
  for (int n = 0; n < 8; ++n) {
    v8f c = vzero;
    for (int kk = 0; kk < 4; ++kk)
      c = wmma_bf16(aq[kk], load_b_frag(ktv + kk*32*128 + n*16, 128), c);
    const int nn = n*16 + nl;
    for (int v = 0; v < 8; ++v)
      ls_x[(mb + v) * 128 + nn] += c[v];
  }
  wave_lds_fence();

  // ---- LN2 -> ls_h ----
  for (int r = 0; r < 16; ++r) {
    float4 vv = *(const float4*)&ls_x[(r0 + r) * 128 + lane * 4];
    float mean, inv; wave_stats(vv, mean, inv);
    const int c0 = lane * 4;
    v4bf h;
    h[0] = (bf16_t)((vv.x - mean) * inv * g2[c0+0] + b2[c0+0]);
    h[1] = (bf16_t)((vv.y - mean) * inv * g2[c0+1] + b2[c0+1]);
    h[2] = (bf16_t)((vv.z - mean) * inv * g2[c0+2] + b2[c0+2]);
    h[3] = (bf16_t)((vv.w - mean) * inv * g2[c0+3] + b2[c0+3]);
    *(v4bf*)&ls_h[(r0 + r) * 128 + c0] = h;
  }
  wave_lds_fence();

  // ---- FFN streamed over FF in chunks of 32 ----
  for (int kk = 0; kk < 4; ++kk) ah[kk] = load_a_frag(&ls_h[r0*128 + kk*32], 128);
  v8f acc[8];
  for (int n = 0; n < 8; ++n) acc[n] = vzero;
  bf16_t* myu = &ls_u[wave][0];

  for (int ch = 0; ch < 16; ++ch) {
    for (int t = 0; t < 2; ++t) {
      v8f c = vzero;
      for (int kk = 0; kk < 4; ++kk)
        c = wmma_bf16(ah[kk], load_b_frag(w1 + kk*32*512 + ch*32 + t*16, 512), c);
      const int ncol = t*16 + nl;
      const float bb = bias1[ch*32 + ncol];
      for (int v = 0; v < 8; ++v) {
        float u = c[v] + bb;
        u = u > 0.0f ? u : 0.0f;
        myu[(hl*8 + v) * 32 + ncol] = (bf16_t)u;
      }
    }
    wave_lds_fence();
    v16bf au = load_a_frag(myu, 32);
    for (int n = 0; n < 8; ++n)
      acc[n] = wmma_bf16(au, load_b_frag(w2 + ch*32*128 + n*16, 128), acc[n]);
    wave_lds_fence();                      // reads done before next chunk store
  }

  // ---- x += ffn + b2 ; coalesced store ----
  for (int n = 0; n < 8; ++n) {
    const int nn = n*16 + nl;
    const float bb = bias2[nn];
    for (int v = 0; v < 8; ++v)
      ls_x[(mb + v) * 128 + nn] += acc[n][v] + bb;
  }
  wave_lds_fence();
  for (int r = 0; r < 16; ++r)
    *(float4*)(xout + (rowBase + r0 + r) * 128 + lane * 4) =
        *(const float4*)&ls_x[(r0 + r) * 128 + lane * 4];
}

// ---------------------------------------------------------------------------
// Kernel 4: out = x @ out_w + out_b   (fp32 output)
__global__ void __launch_bounds__(256) k_final(
    const float* __restrict__ xin, const bf16_t* __restrict__ ow,
    const float* __restrict__ ob, float* __restrict__ out)
{
  __shared__ __align__(32) bf16_t ls_h[128 * 128];
  const int wave = threadIdx.x >> 5;
  const int lane = threadIdx.x & 31;
  const int r0   = wave * 16;
  const int hl   = lane >> 4, nl = lane & 15;
  const int mb   = r0 + hl * 8;
  const size_t rowBase = (size_t)blockIdx.x * ROWS_PER_TILE;
  const v8f vzero = {0.f,0.f,0.f,0.f,0.f,0.f,0.f,0.f};

  for (int r = 0; r < 16; ++r) {
    float4 vv = ((const float4*)(xin + (rowBase + r0 + r) * 128))[lane];
    const int c0 = lane * 4;
    v4bf h; h[0]=(bf16_t)vv.x; h[1]=(bf16_t)vv.y; h[2]=(bf16_t)vv.z; h[3]=(bf16_t)vv.w;
    *(v4bf*)&ls_h[(r0 + r) * 128 + c0] = h;
  }
  wave_lds_fence();

  v16bf a[4];
  for (int kk = 0; kk < 4; ++kk) a[kk] = load_a_frag(&ls_h[r0*128 + kk*32], 128);
  for (int n = 0; n < 8; ++n) {
    v8f c = vzero;
    for (int kk = 0; kk < 4; ++kk)
      c = wmma_bf16(a[kk], load_b_frag(ow + kk*32*128 + n*16, 128), c);
    const int nn = n*16 + nl;
    const float bb = ob[nn];
    for (int v = 0; v < 8; ++v)
      out[(rowBase + mb + v) * 128 + nn] = c[v] + bb;
  }
}

// ---------------------------------------------------------------------------
extern "C" void kernel_launch(void* const* d_in, const int* in_sizes, int n_in,
                              void* d_out, int out_size, void* d_ws, size_t ws_size,
                              hipStream_t stream)
{
  (void)in_sizes; (void)n_in; (void)out_size; (void)ws_size;
  const int Dsq = 128 * 128;       // 16384
  const int W1n = 128 * 512;       // 65536

  char* ws = (char*)d_ws;
  size_t off = 0;
  float*  xbuf  = (float*) (ws + off); off += (size_t)N_ROWS * 128 * sizeof(float);
  float*  parts = (float*) (ws + off); off += (size_t)NBLOCKS1 * Dsq * sizeof(float);
  bf16_t* ktv16 = (bf16_t*)(ws + off); off += (size_t)Dsq * sizeof(bf16_t);
  bf16_t* Kw16  = (bf16_t*)(ws + off); off += (size_t)Dsq * sizeof(bf16_t);
  bf16_t* Qw16  = (bf16_t*)(ws + off); off += (size_t)Dsq * sizeof(bf16_t);
  bf16_t* Vw16  = (bf16_t*)(ws + off); off += (size_t)Dsq * sizeof(bf16_t);
  bf16_t* w1_16 = (bf16_t*)(ws + off); off += (size_t)W1n * sizeof(bf16_t);
  bf16_t* w2_16 = (bf16_t*)(ws + off); off += (size_t)W1n * sizeof(bf16_t);
  bf16_t* ow16  = (bf16_t*)(ws + off); off += (size_t)Dsq * sizeof(bf16_t);

  // Convert weights to bf16 (every call; no cached state).
  k_cvt_bf16<<<64,  256, 0, stream>>>((const float*)d_in[1],  Kw16,  Dsq);
  k_cvt_bf16<<<64,  256, 0, stream>>>((const float*)d_in[2],  Qw16,  Dsq);
  k_cvt_bf16<<<64,  256, 0, stream>>>((const float*)d_in[3],  Vw16,  Dsq);
  k_cvt_bf16<<<256, 256, 0, stream>>>((const float*)d_in[8],  w1_16, W1n);
  k_cvt_bf16<<<256, 256, 0, stream>>>((const float*)d_in[10], w2_16, W1n);
  k_cvt_bf16<<<64,  256, 0, stream>>>((const float*)d_in[12], ow16,  Dsq);

  for (int t = 0; t < T_STEPS; ++t) {
    const float* xt = (t == 0) ? (const float*)d_in[0] : xbuf;
    k_ln1_ktv<<<NBLOCKS1, 256, 0, stream>>>(
        xt, Kw16, Vw16,
        (const float*)d_in[4], (const float*)d_in[5], parts);
    k_ktv_reduce<<<Dsq / 1024, 256, 0, stream>>>(parts, ktv16);
    k_attn_ffn<<<NBLOCKS3, 256, 0, stream>>>(
        xt, xbuf, Qw16, ktv16,
        (const float*)d_in[4], (const float*)d_in[5],
        (const float*)d_in[6], (const float*)d_in[7],
        w1_16, (const float*)d_in[9], w2_16, (const float*)d_in[11]);
  }
  k_final<<<NBLOCKS3, 256, 0, stream>>>(xbuf, ow16, (const float*)d_in[13],
                                        (float*)d_out);
}
